// BasicLayer_up_83923660964627
// MI455X (gfx1250) — compile-verified, matrix operand order checked by reading, not run on hardware
//
#include <hip/hip_runtime.h>

typedef __attribute__((ext_vector_type(16))) _Float16 v16h;
typedef __attribute__((ext_vector_type(8)))  float    v8f;

#define EPI_NONE     0
#define EPI_SOFTPLUS 1

#define HW_   32
#define L_    1024
#define D_    384
#define DI_   768
#define E_    1536
#define XDB_  56
#define DTR_  24
#define NST_  16
#define B_    4
#define M_    4096   // B_ * L_

// grid index corresponding to sequence position p for scan direction `dir`
__device__ __forceinline__ int grid_of_seq(int dir, int p) {
  switch (dir) {
    case 0: return p;
    case 1: { int w = p >> 5; int h = 31 - (p & 31); return h * 32 + w; }
    case 2: return 1023 - p;
    default: { int h = p & 31; int w = 31 - (p >> 5); return h * 32 + w; }
  }
}

__device__ __forceinline__ long row_of(int m, int dir) {
  if (dir < 0) return m;
  const int b = m >> 10, l = m & 1023;
  return (long)(b << 10) + grid_of_seq(dir, l);
}

__device__ __forceinline__ float sigmoidf_(float x) { return 1.f / (1.f + __expf(-x)); }

// Load one 16x32 (or 32x16^T) f16 fragment for this lane from row pointer p.
// ISA 7.12.2 16-bit layout: lane half `hi` holds K runs {8hi..8hi+7} and
// {16+8hi..16+8hi+7} -> two contiguous 8-float runs -> 4x global_load_b128.
__device__ __forceinline__ v16h load_frag(const float* __restrict__ p, int hi) {
  const float4* q0 = (const float4*)(p + 8 * hi);
  const float4* q1 = (const float4*)(p + 16 + 8 * hi);
  const float4 x0 = q0[0], x1 = q0[1], x2 = q1[0], x3 = q1[1];
  v16h r;
  r[0]  = (_Float16)x0.x; r[1]  = (_Float16)x0.y; r[2]  = (_Float16)x0.z; r[3]  = (_Float16)x0.w;
  r[4]  = (_Float16)x1.x; r[5]  = (_Float16)x1.y; r[6]  = (_Float16)x1.z; r[7]  = (_Float16)x1.w;
  r[8]  = (_Float16)x2.x; r[9]  = (_Float16)x2.y; r[10] = (_Float16)x2.z; r[11] = (_Float16)x2.w;
  r[12] = (_Float16)x3.x; r[13] = (_Float16)x3.y; r[14] = (_Float16)x3.z; r[15] = (_Float16)x3.w;
  return r;
}

// Epilogue for one 16x16 tile (C/D layout: VGPR r -> M=r (+8 for hi lanes), N=lo)
__device__ __forceinline__ void store_tile(const v8f& acc, int m_base, int n_base,
                                           int lo, int hi,
                                           const float* __restrict__ bias,
                                           const float* __restrict__ addsrc,
                                           float* __restrict__ out, int ldo,
                                           int N, int out_dir, int accumulate, int epi) {
  const int n_out = n_base + lo;
  if (n_out >= N) return;                      // after all WMMAs: divergence OK
  const float bval = bias ? bias[n_out] : 0.f;
#pragma unroll
  for (int r = 0; r < 8; ++r) {
    const int m = m_base + r + 8 * hi;
    float v = acc[r] + bval;
    if (epi == EPI_SOFTPLUS) v = (v > 20.f) ? v : log1pf(__expf(v));
    if (addsrc) v += addsrc[(long)m * ldo + n_out];
    float* p = out + row_of(m, out_dir) * (long)ldo + n_out;
    if (accumulate) *p += v; else *p = v;      // scatter map is a bijection: no races
  }
}

// ---------------------------------------------------------------------------
// WMMA GEMM:  out[m,n] (+)= epi( sum_k A(row(m),k) * W[n,k] + bias[n] ) (+addsrc)
// Requirements: M % 32 == 0, K % 32 == 0 (caller pads), W rows clamp-indexed.
// One wave computes a 32x32 block = 2x2 WMMA tiles (4 wmma per K-step).
// ---------------------------------------------------------------------------
__global__ void gemm_wmma(const float* __restrict__ A, int lda,
                          const float* __restrict__ W,
                          const float* __restrict__ bias,
                          const float* __restrict__ addsrc,
                          float* __restrict__ out, int ldo,
                          int M, int N, int K,
                          int in_dir, int out_dir, int accumulate, int epi) {
  const int wave = threadIdx.x >> 5;
  const int lane = threadIdx.x & 31;
  const int sb_n = (N + 31) >> 5;
  const int sb_m = M >> 5;
  const int sb = blockIdx.x * (blockDim.x >> 5) + wave;
  if (sb >= sb_m * sb_n) return;               // wave-uniform: EXEC stays full
  const int bm = (sb / sb_n) << 5;
  const int bn = (sb % sb_n) << 5;
  const int hi = lane >> 4, lo = lane & 15;

  const float* Ar0 = A + row_of(bm + lo,      in_dir) * lda;
  const float* Ar1 = A + row_of(bm + 16 + lo, in_dir) * lda;
  const int nb0 = bn + lo, nb1 = bn + 16 + lo;
  const float* Wr0 = W + (long)(nb0 < N ? nb0 : 0) * K;   // clamp: loads stay legal,
  const float* Wr1 = W + (long)(nb1 < N ? nb1 : 0) * K;   // stores are guarded

  v8f acc00 = {}, acc01 = {}, acc10 = {}, acc11 = {};
  for (int k0 = 0; k0 < K; k0 += 32) {
    if (k0 + 32 < K) {                          // gfx1250 global_prefetch_b8
      __builtin_prefetch(Ar0 + k0 + 32, 0, 1);
      __builtin_prefetch(Wr0 + k0 + 32, 0, 1);
    }
    const v16h a0 = load_frag(Ar0 + k0, hi);
    const v16h a1 = load_frag(Ar1 + k0, hi);
    const v16h b0 = load_frag(Wr0 + k0, hi);
    const v16h b1 = load_frag(Wr1 + k0, hi);
    acc00 = __builtin_amdgcn_wmma_f32_16x16x32_f16(false, a0, false, b0, (short)0, acc00, false, false);
    acc01 = __builtin_amdgcn_wmma_f32_16x16x32_f16(false, a0, false, b1, (short)0, acc01, false, false);
    acc10 = __builtin_amdgcn_wmma_f32_16x16x32_f16(false, a1, false, b0, (short)0, acc10, false, false);
    acc11 = __builtin_amdgcn_wmma_f32_16x16x32_f16(false, a1, false, b1, (short)0, acc11, false, false);
  }
  store_tile(acc00, bm,      bn,      lo, hi, bias, addsrc, out, ldo, N, out_dir, accumulate, epi);
  store_tile(acc01, bm,      bn + 16, lo, hi, bias, addsrc, out, ldo, N, out_dir, accumulate, epi);
  store_tile(acc10, bm + 16, bn,      lo, hi, bias, addsrc, out, ldo, N, out_dir, accumulate, epi);
  store_tile(acc11, bm + 16, bn + 16, lo, hi, bias, addsrc, out, ldo, N, out_dir, accumulate, epi);
}

// ---------------------------------------------------------------------------
// Zero-padded repack: dst[r, 0:dst_ld] = { src[r, 0:ncopy], 0... }
// ---------------------------------------------------------------------------
__global__ void pack_pad(const float* __restrict__ src, int src_ld, int ncopy,
                         float* __restrict__ dst, int dst_ld, int rows) {
  const long idx = (long)blockIdx.x * blockDim.x + threadIdx.x;
  if (idx >= (long)rows * dst_ld) return;
  const int c = (int)(idx % dst_ld);
  const long r = idx / dst_ld;
  dst[idx] = (c < ncopy) ? src[r * src_ld + c] : 0.f;
}

// ---------------------------------------------------------------------------
// Depthwise causal conv (D_CONV=4) + bias + SiLU.  Reads u-half of xz.
// ---------------------------------------------------------------------------
__global__ void conv_silu(const float* __restrict__ xz,
                          const float* __restrict__ cw,
                          const float* __restrict__ cb,
                          float* __restrict__ uc) {
  const long idx = (long)blockIdx.x * blockDim.x + threadIdx.x;
  if (idx >= (long)M_ * DI_) return;
  const int d = (int)(idx % DI_);
  const long m = idx / DI_;
  const int l = (int)(m & 1023);
  const long bbase = m - l;
  float s = cb[d];
#pragma unroll
  for (int j = 0; j < 4; ++j) {
    const int ls = l - 3 + j;
    if (ls >= 0) s += xz[(bbase + ls) * E_ + d] * cw[d * 4 + j];
  }
  uc[idx] = s * sigmoidf_(s);
}

// ---------------------------------------------------------------------------
// Selective scan: one thread per (batch, inner-channel); 16 states in VGPRs;
// B/C vectors broadcast via LDS; fused D-skip + SiLU(z) gate.
// ---------------------------------------------------------------------------
__global__ void scan_kernel(const float* __restrict__ dt,
                            const float* __restrict__ uc,
                            const float* __restrict__ xdb,
                            const float* __restrict__ xz,
                            const float* __restrict__ A_log,
                            const float* __restrict__ Dp,
                            float* __restrict__ y) {
  const int b = blockIdx.x / (DI_ / 256);
  const int d = (blockIdx.x % (DI_ / 256)) * 256 + threadIdx.x;
  float A[NST_], h[NST_];
#pragma unroll
  for (int n = 0; n < NST_; ++n) { A[n] = -__expf(A_log[d * NST_ + n]); h[n] = 0.f; }
  const float Dd = Dp[d];
  __shared__ float sBC[2 * NST_];
  for (int l = 0; l < L_; ++l) {
    const long m = ((long)b << 10) + l;
    __syncthreads();
    if (threadIdx.x < 2 * NST_) sBC[threadIdx.x] = xdb[m * XDB_ + DTR_ + threadIdx.x];
    __syncthreads();
    const float dtv = dt[m * DI_ + d];
    const float uv  = uc[m * DI_ + d];
    const float du  = dtv * uv;
    float yv = 0.f;
#pragma unroll
    for (int n = 0; n < NST_; ++n) {
      h[n] = h[n] * __expf(dtv * A[n]) + du * sBC[n];
      yv += h[n] * sBC[NST_ + n];
    }
    yv += uv * Dd;
    const float zv = xz[m * E_ + DI_ + d];
    yv *= zv * sigmoidf_(zv);
    y[m * DI_ + d] = yv;
  }
}

// ---------------------------------------------------------------------------
// LayerNorm over last dim D; one 256-thread block per row; LDS tree reduce.
// ---------------------------------------------------------------------------
__global__ void layernorm_kernel(const float* __restrict__ in,
                                 const float* __restrict__ w,
                                 const float* __restrict__ b,
                                 float* __restrict__ out, int D) {
  const long row = blockIdx.x;
  const float* x = in + row * D;
  float* o = out + row * D;
  float s = 0.f, s2 = 0.f;
  for (int i = threadIdx.x; i < D; i += blockDim.x) { float v = x[i]; s += v; s2 += v * v; }
  __shared__ float shs[256], shs2[256];
  shs[threadIdx.x] = s; shs2[threadIdx.x] = s2;
  __syncthreads();
  for (int off = 128; off > 0; off >>= 1) {
    if (threadIdx.x < off) { shs[threadIdx.x] += shs[threadIdx.x + off];
                             shs2[threadIdx.x] += shs2[threadIdx.x + off]; }
    __syncthreads();
  }
  const float mu  = shs[0] / D;
  const float var = shs2[0] / D - mu * mu;
  const float inv = rsqrtf(var + 1e-5f);
  for (int i = threadIdx.x; i < D; i += blockDim.x)
    o[i] = (x[i] - mu) * inv * w[i] + b[i];
}

// ---------------------------------------------------------------------------
// Pixel shuffle: (B,32,32,768) -> (B,64,64,192)
// ---------------------------------------------------------------------------
__global__ void pixel_shuffle(const float* __restrict__ xe, float* __restrict__ out) {
  const long idx = (long)blockIdx.x * blockDim.x + threadIdx.x;
  if (idx >= (long)B_ * 64 * 64 * 192) return;
  const int c = (int)(idx % 192);
  long t = idx / 192;
  const int W = (int)(t % 64); t /= 64;
  const int H = (int)(t % 64); t /= 64;
  const int b = (int)t;
  const int h = H >> 1, s1 = H & 1, w = W >> 1, s2 = W & 1;
  out[idx] = xe[(((long)b * L_ + h * 32 + w) * (2 * D_)) + (s1 * 2 + s2) * 192 + c];
}

// ---------------------------------------------------------------------------
static inline void launch_gemm(const float* A, int lda, const float* W,
                               const float* bias, const float* addsrc,
                               float* out, int ldo, int M, int N, int K,
                               int in_dir, int out_dir, int accumulate, int epi,
                               hipStream_t stream) {
  const int sbs = (M >> 5) * ((N + 31) >> 5);
  const int blocks = (sbs + 7) / 8;   // 8 waves / block, one 32x32 block / wave
  gemm_wmma<<<blocks, 256, 0, stream>>>(A, lda, W, bias, addsrc, out, ldo,
                                        M, N, K, in_dir, out_dir, accumulate, epi);
}

extern "C" void kernel_launch(void* const* d_in, const int* in_sizes, int n_in,
                              void* d_out, int out_size, void* d_ws, size_t ws_size,
                              hipStream_t stream) {
  (void)in_sizes; (void)n_in; (void)out_size; (void)ws_size;
  const float* x      = (const float*)d_in[0];
  const float* in_w   = (const float*)d_in[1];   // (2,1536,384)
  const float* conv_w = (const float*)d_in[2];   // (2,768,4)
  const float* conv_b = (const float*)d_in[3];   // (2,768)
  const float* xp_w   = (const float*)d_in[4];   // (2,56,768)
  const float* dt_w   = (const float*)d_in[5];   // (2,768,24)
  const float* dt_b   = (const float*)d_in[6];   // (2,768)
  const float* A_log  = (const float*)d_in[7];   // (2,768,16)
  const float* Dp     = (const float*)d_in[8];   // (2,768)
  const float* mnw    = (const float*)d_in[9];   // (2,384)
  const float* mnb    = (const float*)d_in[10];
  const float* mout_w = (const float*)d_in[11];  // (2,384,768)
  const float* bpw    = (const float*)d_in[12];  // (2,384,384)
  const float* bpb    = (const float*)d_in[13];
  const float* lnw    = (const float*)d_in[14];
  const float* lnb    = (const float*)d_in[15];
  const float* exp_w  = (const float*)d_in[16];  // (768,384)
  const float* penw   = (const float*)d_in[17];  // (192)
  const float* penb   = (const float*)d_in[18];

  float* ws    = (float*)d_ws;
  float* xz    = ws;                              // M_*E_   = 6,291,456
  float* uc    = xz    + (long)M_ * E_;           // M_*DI_  = 3,145,728
  float* xdbuf = uc    + (long)M_ * DI_;          // M_*XDB_ =   229,376
  float* dtbuf = xdbuf + (long)M_ * XDB_;         // M_*DI_  = 3,145,728
  float* ybuf  = dtbuf + (long)M_ * DI_;          // M_*DI_  = 3,145,728
  float* yacc  = ybuf  + (long)M_ * DI_;          // M_*D_   = 1,572,864
  float* xnext = yacc  + (long)M_ * D_;           // M_*D_
  float* xcur  = xnext + (long)M_ * D_;           // M_*D_
  float* xdb24 = xcur  + (long)M_ * D_;           // M_*32   =   131,072
  float* dtwp  = xdb24 + (long)M_ * 32;           // DI_*32  =    24,576  (~84 MB total)
  float* lnbuf = ybuf;   // dead-buffer alias (ybuf consumed before LN of yacc)
  float* xe    = xz;     // final-stage aliases
  float* shuf  = uc;

  const float* xin = x;
  for (int i = 0; i < 2; ++i) {
    (void)hipMemsetAsync(yacc, 0, (size_t)M_ * D_ * sizeof(float), stream);
    const float* in_w_l = in_w   + (long)i * E_ * D_;
    const float* cw_l   = conv_w + (long)i * DI_ * 4;
    const float* cb_l   = conv_b + (long)i * DI_;
    const float* xp_l   = xp_w   + (long)i * XDB_ * DI_;
    const float* dtw_l  = dt_w   + (long)i * DI_ * DTR_;
    const float* dtb_l  = dt_b   + (long)i * DI_;
    const float* Al_l   = A_log  + (long)i * DI_ * NST_;
    const float* Dp_l   = Dp     + (long)i * DI_;
    const float* mout_l = mout_w + (long)i * D_ * DI_;

    // zero-pad dt_w (768x24 -> 768x32) once per layer
    pack_pad<<<(DI_ * 32 + 255) / 256, 256, 0, stream>>>(dtw_l, DTR_, DTR_, dtwp, 32, DI_);

    for (int dir = 0; dir < 4; ++dir) {
      // xz = gather_dir(x) @ in_w^T                       (4096 x 1536 x 384)
      launch_gemm(xin, D_, in_w_l, nullptr, nullptr, xz, E_,
                  M_, E_, D_, dir, -1, 0, EPI_NONE, stream);
      // depthwise conv + SiLU on u-half
      conv_silu<<<(int)(((long)M_ * DI_ + 255) / 256), 256, 0, stream>>>(xz, cw_l, cb_l, uc);
      // xdb = u @ x_proj^T                                (4096 x 56 x 768)
      launch_gemm(uc, DI_, xp_l, nullptr, nullptr, xdbuf, XDB_,
                  M_, XDB_, DI_, -1, -1, 0, EPI_NONE, stream);
      // zero-pad xdb[:, :24] -> 4096x32
      pack_pad<<<(int)(((long)M_ * 32 + 255) / 256), 256, 0, stream>>>(xdbuf, XDB_, DTR_,
                                                                       xdb24, 32, (int)M_);
      // dt = softplus(xdb24 @ dtwp^T + dt_b)              (4096 x 768 x 32)
      launch_gemm(xdb24, 32, dtwp, dtb_l, nullptr, dtbuf, DI_,
                  M_, DI_, 32, -1, -1, 0, EPI_SOFTPLUS, stream);
      // sequential selective scan, fused skip + SiLU(z) gate
      scan_kernel<<<B_ * (DI_ / 256), 256, 0, stream>>>(dtbuf, uc, xdbuf, xz,
                                                        Al_l, Dp_l, ybuf);
      // yacc[grid] += y @ mout^T  (un-permute scatter)    (4096 x 384 x 768)
      launch_gemm(ybuf, DI_, mout_l, nullptr, nullptr, yacc, D_,
                  M_, D_, DI_, -1, dir, 1, EPI_NONE, stream);
    }
    // LN(yacc) -> lnbuf ; xnext = xin + lnbuf @ bpw^T + bpb ; xcur = LN(xnext)
    layernorm_kernel<<<(int)M_, 256, 0, stream>>>(yacc, mnw + (long)i * D_,
                                                  mnb + (long)i * D_, lnbuf, D_);
    launch_gemm(lnbuf, D_, bpw + (long)i * D_ * D_, bpb + (long)i * D_, xin,
                xnext, D_, M_, D_, D_, -1, -1, 0, EPI_NONE, stream);
    layernorm_kernel<<<(int)M_, 256, 0, stream>>>(xnext, lnw + (long)i * D_,
                                                  lnb + (long)i * D_, xcur, D_);
    xin = xcur;
  }
  // final expansion, pixel shuffle, LN over 192 channels -> d_out
  launch_gemm(xcur, D_, exp_w, nullptr, nullptr, xe, 2 * D_,
              M_, 2 * D_, D_, -1, -1, 0, EPI_NONE, stream);
  pixel_shuffle<<<(int)(((long)B_ * 64 * 64 * 192 + 255) / 256), 256, 0, stream>>>(xe, shuf);
  layernorm_kernel<<<B_ * 64 * 64, 256, 0, stream>>>(shuf, penw, penb,
                                                     (float*)d_out, 192);
}
